// NeuralIF_13786845020472
// MI455X (gfx1250) — compile-verified, matrix operand order checked by reading, not run on hardware
//
#include <hip/hip_runtime.h>

typedef __attribute__((ext_vector_type(16))) _Float16 v16h;
typedef __attribute__((ext_vector_type(8)))  _Float16 v8h;
typedef __attribute__((ext_vector_type(2)))  _Float16 v2h;
typedef __attribute__((ext_vector_type(8)))  float    v8f;

#define NNODES 100000
#define NEDGES 3200000
#define FN 8
#define FE 8
#define HID 32
#define WAVES_PER_BLOCK 8

// Lane's 16 halves of a 32x16 f16 B tile, standard ISA K mapping:
// half j -> K = grp*8 + j ; half 8+j -> K = 16 + grp*8 + j.
__device__ inline v16h load_B_tile(const float* __restrict__ W, int fan_in,
                                   int fan_out, int col, int grp) {
  v16h b;
#pragma unroll
  for (int j = 0; j < 8; ++j) {
    int k0 = grp * 8 + j;
    int k1 = 16 + grp * 8 + j;
    float w0 = (col < fan_out && k0 < fan_in) ? W[k0 * fan_out + col] : 0.0f;
    float w1 = (col < fan_out && k1 < fan_in) ? W[k1 * fan_out + col] : 0.0f;
    b[j]     = (_Float16)w0;
    b[8 + j] = (_Float16)w1;
  }
  return b;
}

// Layer-2 B tile with the pair-interleaved K permutation used by the LDS
// staging buffer: position (grp, half j') holds feature
//   k = 8*grp + (j'>>1) + 16*(j'&1)
// (a bijection onto the same K set the hardware assigns to this lane group,
//  so the WMMA dot product is unchanged as long as A uses the same order).
__device__ inline v16h load_B_tile_perm(const float* __restrict__ W,
                                        int fan_out, int col, int grp) {
  v16h b;
#pragma unroll
  for (int jp = 0; jp < 16; ++jp) {
    int k = 8 * grp + (jp >> 1) + 16 * (jp & 1);
    float w = (col < fan_out) ? W[k * fan_out + col] : 0.0f;
    b[jp] = (_Float16)w;
  }
  return b;
}

__device__ inline void pack8(v16h& a, int base, float4 lo, float4 hi, float s) {
  a[base + 0] = (_Float16)(lo.x * s); a[base + 1] = (_Float16)(lo.y * s);
  a[base + 2] = (_Float16)(lo.z * s); a[base + 3] = (_Float16)(lo.w * s);
  a[base + 4] = (_Float16)(hi.x * s); a[base + 5] = (_Float16)(hi.y * s);
  a[base + 6] = (_Float16)(hi.z * s); a[base + 7] = (_Float16)(hi.w * s);
}

__device__ inline v8f bcast8(float v) {
  v8f r;
#pragma unroll
  for (int i = 0; i < 8; ++i) r[i] = v;
  return r;
}

__global__ void zero_ws_kernel(float* __restrict__ p, long long count) {
  long long i = (long long)blockIdx.x * blockDim.x + threadIdx.x;
  long long stride = (long long)gridDim.x * blockDim.x;
  for (; i < count; i += stride) p[i] = 0.0f;
}

// ---------------------------------------------------------------------------
// Edge kernel: per wave, one 16-edge tile per iteration.
//   A  = [x[row] | x[col] | edge_attr | 0pad]  (16 x 32 f16)
//   L1 = 2x v_wmma_f32_16x16x32_f16 (bias preloaded in C) + ReLU
//   LDS: pair-interleaved f16 hidden tile, 8x ds_store_b32 / 2x ds_load_b128
//   L2 = 1x v_wmma with K-permuted B2 (bias preloaded in C)
//   store edge_embedding, atomic scatter-add to agg/cnt over source node.
// ---------------------------------------------------------------------------
__global__ void __launch_bounds__(256)
edge_mlp_kernel(const float* __restrict__ x, const int* __restrict__ ei,
                const float* __restrict__ ea,
                const float* __restrict__ We1, const float* __restrict__ be1,
                const float* __restrict__ We2, const float* __restrict__ be2,
                float* __restrict__ edge_out,
                float* __restrict__ agg, float* __restrict__ cnt) {
  __shared__ __attribute__((aligned(32)))
      _Float16 hid[WAVES_PER_BLOCK][16][HID];  // [row][2*n + sel]

  const int lane = threadIdx.x & 31;
  const int wv   = threadIdx.x >> 5;
  const int grp  = lane >> 4;
  const int n    = lane & 15;  // output column
  const int m    = lane & 15;  // A-matrix row
  const int nwaves = gridDim.x * (blockDim.x >> 5);
  const int gw     = blockIdx.x * (blockDim.x >> 5) + wv;

  // Wave-invariant operands
  const v16h B1a = load_B_tile(We1, 2 * FN + FE, HID, n,      grp);
  const v16h B1b = load_B_tile(We1, 2 * FN + FE, HID, n + 16, grp);
  const v16h B2  = load_B_tile_perm(We2, FE, n, grp);
  const v8f  Cb1a = bcast8(be1[n]);
  const v8f  Cb1b = bcast8(be1[16 + n]);
  const v8f  Cb2  = bcast8((n < FE) ? be2[n] : 0.0f);

  const int ntiles = NEDGES / 16;
  for (int tile = gw; tile < ntiles; tile += nwaves) {
    const int e = tile * 16 + m;
    const int node = grp ? ei[NEDGES + e] : ei[e];
    const float4* xp = (const float4*)(x + (long long)node * FN);
    float4 x0 = xp[0], x1 = xp[1];
    v16h A;
    pack8(A, 0, x0, x1, 1.0f);
    if (grp == 0) {  // K 16..23 = edge_attr
      const float4* ep = (const float4*)(ea + (long long)e * FE);
      pack8(A, 8, ep[0], ep[1], 1.0f);
    } else {         // K 24..31 zero pad
#pragma unroll
      for (int j = 8; j < 16; ++j) A[j] = (_Float16)0.0f;
    }

    v8f c0 = __builtin_amdgcn_wmma_f32_16x16x32_f16(false, A, false, B1a,
                                                    (short)0, Cb1a, false, false);
    v8f c1 = __builtin_amdgcn_wmma_f32_16x16x32_f16(false, A, false, B1b,
                                                    (short)0, Cb1b, false, false);
    // ReLU + pack (feature n, feature 16+n) pairs into dword LDS stores
#pragma unroll
    for (int v = 0; v < 8; ++v) {
      const int row = v + 8 * grp;
      v2h p;
      p[0] = (_Float16)fmaxf(c0[v], 0.0f);
      p[1] = (_Float16)fmaxf(c1[v], 0.0f);
      *(v2h*)&hid[wv][row][2 * n] = p;
    }
    asm volatile("s_wait_dscnt 0" ::: "memory");  // wave-level LDS fence

    // 16 contiguous halves = permuted-K A operand for layer 2
    v8h lo = *(const v8h*)&hid[wv][m][16 * grp];
    v8h hi = *(const v8h*)&hid[wv][m][16 * grp + 8];
    v16h A2;
#pragma unroll
    for (int j = 0; j < 8; ++j) { A2[j] = lo[j]; A2[8 + j] = hi[j]; }

    v8f c2 = __builtin_amdgcn_wmma_f32_16x16x32_f16(false, A2, false, B2,
                                                    (short)0, Cb2, false, false);
    if (n < FE) {
#pragma unroll
      for (int v = 0; v < 8; ++v) {
        const int row = v + 8 * grp;
        const int ee  = tile * 16 + row;
        const float val = c2[v];
        edge_out[(long long)ee * FE + n] = val;
        const int dst = ei[ee];  // scatter over source node (row)
        atomicAdd(&agg[(long long)dst * FE + n], val);
        if (n == 0) atomicAdd(&cnt[dst], 1.0f);
      }
    }
  }
}

// ---------------------------------------------------------------------------
// Node kernel: A = [x[i] | agg[i]/max(cnt,1) | 0pad], same 3-WMMA structure.
// ---------------------------------------------------------------------------
__global__ void __launch_bounds__(256)
node_mlp_kernel(const float* __restrict__ x, const float* __restrict__ agg,
                const float* __restrict__ cnt,
                const float* __restrict__ Wn1, const float* __restrict__ bn1,
                const float* __restrict__ Wn2, const float* __restrict__ bn2,
                float* __restrict__ node_out) {
  __shared__ __attribute__((aligned(32)))
      _Float16 hid[WAVES_PER_BLOCK][16][HID];

  const int lane = threadIdx.x & 31;
  const int wv   = threadIdx.x >> 5;
  const int grp  = lane >> 4;
  const int n    = lane & 15;
  const int m    = lane & 15;
  const int nwaves = gridDim.x * (blockDim.x >> 5);
  const int gw     = blockIdx.x * (blockDim.x >> 5) + wv;

  const v16h B1a = load_B_tile(Wn1, FN + FE, HID, n,      grp);
  const v16h B1b = load_B_tile(Wn1, FN + FE, HID, n + 16, grp);
  const v16h B2  = load_B_tile_perm(Wn2, FN, n, grp);
  const v8f  Cb1a = bcast8(bn1[n]);
  const v8f  Cb1b = bcast8(bn1[16 + n]);
  const v8f  Cb2  = bcast8((n < FN) ? bn2[n] : 0.0f);

  const int ntiles = NNODES / 16;
  for (int tile = gw; tile < ntiles; tile += nwaves) {
    const int i = tile * 16 + m;
    v16h A;
    if (grp == 0) {  // features 0..7 = x[i]
      const float4* xp = (const float4*)(x + (long long)i * FN);
      pack8(A, 0, xp[0], xp[1], 1.0f);
    } else {         // features 8..15 = mean aggregation
      const float inv = 1.0f / fmaxf(cnt[i], 1.0f);
      const float4* ap = (const float4*)(agg + (long long)i * FE);
      pack8(A, 0, ap[0], ap[1], inv);
    }
#pragma unroll
    for (int j = 8; j < 16; ++j) A[j] = (_Float16)0.0f;  // K 16..31 pad

    v8f c0 = __builtin_amdgcn_wmma_f32_16x16x32_f16(false, A, false, B1a,
                                                    (short)0, Cb1a, false, false);
    v8f c1 = __builtin_amdgcn_wmma_f32_16x16x32_f16(false, A, false, B1b,
                                                    (short)0, Cb1b, false, false);
#pragma unroll
    for (int v = 0; v < 8; ++v) {
      const int row = v + 8 * grp;
      v2h p;
      p[0] = (_Float16)fmaxf(c0[v], 0.0f);
      p[1] = (_Float16)fmaxf(c1[v], 0.0f);
      *(v2h*)&hid[wv][row][2 * n] = p;
    }
    asm volatile("s_wait_dscnt 0" ::: "memory");

    v8h lo = *(const v8h*)&hid[wv][m][16 * grp];
    v8h hi = *(const v8h*)&hid[wv][m][16 * grp + 8];
    v16h A2;
#pragma unroll
    for (int j = 0; j < 8; ++j) { A2[j] = lo[j]; A2[8 + j] = hi[j]; }

    v8f c2 = __builtin_amdgcn_wmma_f32_16x16x32_f16(false, A2, false, B2,
                                                    (short)0, Cb2, false, false);
    if (n < FN) {
#pragma unroll
      for (int v = 0; v < 8; ++v) {
        const int row = v + 8 * grp;
        node_out[(long long)(tile * 16 + row) * FN + n] = c2[v];
      }
    }
  }
}

extern "C" void kernel_launch(void* const* d_in, const int* in_sizes, int n_in,
                              void* d_out, int out_size, void* d_ws,
                              size_t ws_size, hipStream_t stream) {
  const float* x   = (const float*)d_in[0];
  const int*   ei  = (const int*)d_in[1];
  const float* ea  = (const float*)d_in[2];
  const float* We1 = (const float*)d_in[3];
  const float* be1 = (const float*)d_in[4];
  const float* We2 = (const float*)d_in[5];
  const float* be2 = (const float*)d_in[6];
  const float* Wn1 = (const float*)d_in[7];
  const float* bn1 = (const float*)d_in[8];
  const float* Wn2 = (const float*)d_in[9];
  const float* bn2 = (const float*)d_in[10];

  float* edge_out = (float*)d_out;                       // [E, FE]
  float* node_out = edge_out + (long long)NEDGES * FE;   // [N, FN]
  float* agg = (float*)d_ws;                             // [N, FE] sums
  float* cnt = agg + (long long)NNODES * FE;             // [N] degrees

  zero_ws_kernel<<<256, 256, 0, stream>>>(agg, (long long)NNODES * (FE + 1));
  edge_mlp_kernel<<<2048, 256, 0, stream>>>(x, ei, ea, We1, be1, We2, be2,
                                            edge_out, agg, cnt);
  node_mlp_kernel<<<784, 256, 0, stream>>>(x, agg, cnt, Wn1, bn1, Wn2, bn2,
                                           node_out);
}
